// RNN_Encoder_40346922779401
// MI455X (gfx1250) — matrix operand order repaired and checked by reading, not compiled, not run on hardware
//
#include <hip/hip_runtime.h>

// ---------------------------------------------------------------------------
// Bidirectional LSTM encoder for MI455X (gfx1250, wave32, WMMA).
//   VOCAB=30000, EMBED=256, HID=128, B=256, T=512, G=4H=512
// Kernel 1: embedding gather + x@W+b  (f16 WMMA, f32 accum) -> xW (f16, ws)
// Kernel 2: persistent recurrent scan; U resident in LDS (f16 frags),
//           xW tiles double-buffered via GLOBAL_LOAD_ASYNC_TO_LDS_B128,
//           z = xW + h@U via v_wmma_f32_16x16x32_f16, gates in f32.
// B fragments are batch-preloaded so ds_load clauses overlap the WMMA chain
// (partial s_wait_dscnt instead of a full stall per WMMA).
// ---------------------------------------------------------------------------

#define DEVINL __device__ __forceinline__

typedef __attribute__((ext_vector_type(16))) _Float16 v16h;
typedef __attribute__((ext_vector_type(8)))  float    v8f;
typedef __attribute__((ext_vector_type(4)))  float    v4f;

constexpr int EMBED = 256;
constexpr int HID   = 128;
constexpr int BATCH = 256;
constexpr int SEQ   = 512;
constexpr int G     = 4 * HID;   // 512 gate columns per direction
#define ZLD (512 + 4)            // padded z-buffer row stride (floats)

// --- CDNA5 WMMA 16-bit fragment position helpers (ISA 7.12.2) --------------
DEVINL int afrag_pos(int m, int k) {
  int kc = k >> 5, ko = k & 31;
  int half = ko >> 4;
  int j = ko & 15;
  return ((kc * 32) + half * 16 + m) * 16 + j;
}
DEVINL int bfrag_pos(int nInTile, int k) {
  int kc = k >> 5;
  int half = (k >> 4) & 1;
  int j = k & 15;
  return (kc * 32 + half * 16 + nInTile) * 16 + j;
}

DEVINL float fast_sigmoid(float x) { return 1.0f / (1.0f + __expf(-x)); }

// wave-relative LDS byte offset for async-LDS instructions
DEVINL unsigned lds_addr32(const void* p) {
  return (unsigned)(unsigned long long)(const __attribute__((address_space(3))) void*)p;
}

// ===========================================================================
// Kernel 1: fused embedding gather + input projection (both directions).
//   grid = (512, 4); block = 256 threads (8 waves); wave -> 2 N-tiles of 16.
// ===========================================================================
__global__ void __launch_bounds__(256)
embed_project_kernel(const int*   __restrict__ tokens,
                     const float* __restrict__ emb,
                     const float* __restrict__ W_fw, const float* __restrict__ b_fw,
                     const float* __restrict__ W_bw, const float* __restrict__ b_bw,
                     _Float16* __restrict__ xWf, _Float16* __restrict__ xWb)
{
  extern __shared__ char smem[];
  _Float16* Wfrag = (_Float16*)smem;                          // 16 nt * 8 kc * 32 * 16
  _Float16* Afrag = (_Float16*)(smem + 16 * 8 * 32 * 16 * 2); //  8 kc * 32 * 16
  int*      sTok  = (int*)(smem + 16 * 8 * 32 * 16 * 2 + 8 * 32 * 16 * 2);

  const int tid  = threadIdx.x;
  const int wave = tid >> 5;
  const int lane = tid & 31;

  const int cg      = blockIdx.y;     // 0..3
  const int dirSel  = cg >> 1;        // 0 fw, 1 bw
  const int colBase = (cg & 1) * 256; // within this direction's 512 cols
  const float* Wsrc = dirSel ? W_bw : W_fw;
  const float* bsrc = dirSel ? b_bw : b_fw;
  _Float16*    dst  = dirSel ? xWb : xWf;

  // ---- stage W block [K=256][N=256] as f16 fragments (once per WG) --------
  for (int i = 0; i < 64; ++i) {
    int e   = tid + i * 256;          // float4 index over [256 k][64 n4]
    int k   = e >> 6;
    int nl0 = (e & 63) * 4;
    v4f w4 = *(const v4f*)(Wsrc + (size_t)k * G + colBase + nl0);
#pragma unroll
    for (int c = 0; c < 4; ++c) {
      int nl = nl0 + c;
      Wfrag[(nl >> 4) * 4096 + bfrag_pos(nl & 15, k)] = (_Float16)w4[c];
    }
  }

  const int nt0 = wave * 2;
  const float bias0 = bsrc[colBase + nt0 * 16 + (lane & 15)];
  const float bias1 = bsrc[colBase + nt0 * 16 + 16 + (lane & 15)];

  for (int rt = 0; rt < 16; ++rt) {
    const int row0 = (blockIdx.x * 16 + rt) * 16;   // flat row = b*T + t
    __syncthreads();                                 // Afrag consumed / Wfrag ready
    if (tid < 16) sTok[tid] = tokens[row0 + tid];
    __syncthreads();

    // ---- gather 16 embedding rows -> f16 A fragments ----------------------
    for (int i = 0; i < 4; ++i) {
      int e  = tid + i * 256;        // float4 over [16 rows][64 k4]
      int r  = e >> 6;
      int k0 = (e & 63) * 4;
      v4f x4 = *(const v4f*)(emb + (size_t)sTok[r] * EMBED + k0);
#pragma unroll
      for (int c = 0; c < 4; ++c)
        Afrag[afrag_pos(r, k0 + c)] = (_Float16)x4[c];
    }
    __syncthreads();

    // ---- 16x16 output tiles over K=256 (8 WMMAs per tile) -----------------
    v16h a[8];
#pragma unroll
    for (int kc = 0; kc < 8; ++kc)
      a[kc] = *(const v16h*)(Afrag + (kc * 32 + lane) * 16);

#pragma unroll
    for (int q = 0; q < 2; ++q) {
      const int nt = nt0 + q;
      // batch-preload all B fragments so the ds_load clause overlaps the
      // WMMA chain (backend retires them with partial s_wait_dscnt)
      v16h b[8];
#pragma unroll
      for (int kc = 0; kc < 8; ++kc)
        b[kc] = *(const v16h*)(Wfrag + nt * 4096 + (kc * 32 + lane) * 16);

      const float bb = q ? bias1 : bias0;
      v8f acc;
#pragma unroll
      for (int r = 0; r < 8; ++r) acc[r] = bb;
#pragma unroll
      for (int kc = 0; kc < 8; ++kc)
        acc = __builtin_amdgcn_wmma_f32_16x16x32_f16(false, a[kc], false, b[kc],
                                                     (short)0, acc, false, false);

      const int col   = colBase + nt * 16 + (lane & 15);
      const int mBase = (lane < 16) ? 0 : 8;
#pragma unroll
      for (int r = 0; r < 8; ++r)
        dst[(size_t)(row0 + mBase + r) * G + col] = (_Float16)acc[r];
    }
  }
}

// ===========================================================================
// Kernel 2: persistent bidirectional LSTM scan with async double-buffering.
//   grid = 32 blocks: blockIdx>>4 = direction, (blockIdx&15)*16 = batch rows.
//   block = 256 threads (8 waves); wave w owns gate cols [w*64, w*64+64).
// ===========================================================================
__global__ void __launch_bounds__(256)
bilstm_scan_kernel(const _Float16* __restrict__ xWf, const _Float16* __restrict__ xWb,
                   const float* __restrict__ U_fw, const float* __restrict__ U_bw,
                   float* __restrict__ out /* [B][T][2H] + hf + cf + hb + cb */)
{
  extern __shared__ char smem[];
  _Float16* Ufrag  = (_Float16*)smem;                           // 32 nt * 4 kc * 32 * 16
  _Float16* hfrag  = (_Float16*)(smem + 32 * 4 * 32 * 16 * 2);  //  4 kc * 32 * 16
  float*    zbuf   = (float*)(smem + 32 * 4 * 32 * 16 * 2 + 4 * 32 * 16 * 2);
  _Float16* xstage = (_Float16*)(smem + 32 * 4 * 32 * 16 * 2 + 4 * 32 * 16 * 2
                                      + 16 * ZLD * 4);          // 2 x [16][512] f16

  const int tid  = threadIdx.x;
  const int wave = tid >> 5;
  const int lane = tid & 31;
  const int dir  = blockIdx.x >> 4;
  const int b0   = (blockIdx.x & 15) * 16;

  const float*    U  = dir ? U_bw : U_fw;
  const _Float16* xW = dir ? xWb : xWf;

  // per-thread slice of the async tile copy: 4 x 16-byte chunks
  const int ar0[4] = { (tid + 0)   >> 6, (tid + 256) >> 6,
                       (tid + 512) >> 6, (tid + 768) >> 6 };
  const int ac0 = (tid & 63) * 8;   // column chunk (halves)

  // ---- kick off async staging of step 0's tile into buffer 0 --------------
  {
    const int s0 = dir ? (SEQ - 1) : 0;
#pragma unroll
    for (int i = 0; i < 4; ++i) {
      const unsigned l = lds_addr32(xstage) + (unsigned)(ar0[i] * 512 + ac0) * 2u;
      const unsigned long long g = (unsigned long long)
          (xW + ((size_t)(b0 + ar0[i]) * SEQ + s0) * G + ac0);
      asm volatile("global_load_async_to_lds_b128 %0, %1, off"
                   :: "v"(l), "v"(g) : "memory");
    }
  }

  // ---- stage U [K=128][N=512] as f16 fragments (resident for whole scan) --
  for (int i = 0; i < 64; ++i) {
    int e  = tid + i * 256;          // float4 over [128 k][128 n4]
    int k  = e >> 7;
    int n0 = (e & 127) * 4;
    v4f u4 = *(const v4f*)(U + (size_t)k * G + n0);
#pragma unroll
    for (int c = 0; c < 4; ++c) {
      int n = n0 + c;
      Ufrag[(n >> 4) * 2048 + bfrag_pos(n & 15, k)] = (_Float16)u4[c];
    }
  }
  for (int i = tid; i < 4 * 32 * 16; i += 256) hfrag[i] = (_Float16)0.0f;

  // per-thread state: row m = tid&15, hidden cols [(tid>>4)*8, +8)
  const int em  = tid & 15;
  const int ec0 = (tid >> 4) * 8;
  float cst[8], hst[8];
#pragma unroll
  for (int e = 0; e < 8; ++e) { cst[e] = 0.0f; hst[e] = 0.0f; }

  const int nt0 = wave * 4;
  int cur = 0;

  for (int it = 0; it < SEQ; ++it) {
    const int s = dir ? (SEQ - 1 - it) : it;
    __syncthreads();   // prev elementwise done: hfrag valid, both buffers safe

    // ---- issue async staging of the NEXT step's tile (other buffer) -------
    if (it + 1 < SEQ) {
      const int sn = dir ? (s - 1) : (s + 1);
      const unsigned lbase = lds_addr32(xstage) + (unsigned)((cur ^ 1) * 8192 * 2);
#pragma unroll
      for (int i = 0; i < 4; ++i) {
        const unsigned l = lbase + (unsigned)(ar0[i] * 512 + ac0) * 2u;
        const unsigned long long g = (unsigned long long)
            (xW + ((size_t)(b0 + ar0[i]) * SEQ + sn) * G + ac0);
        asm volatile("global_load_async_to_lds_b128 %0, %1, off"
                     :: "v"(l), "v"(g) : "memory");
      }
    }

    // ---- z_hU = h @ U via WMMA (store, no RMW) ----------------------------
    v16h a[4];
#pragma unroll
    for (int kc = 0; kc < 4; ++kc)
      a[kc] = *(const v16h*)(hfrag + (kc * 32 + lane) * 16);

#pragma unroll
    for (int q = 0; q < 4; ++q) {
      const int nt = nt0 + q;
      // batch-preload B fragments for this tile (overlap loads with WMMAs)
      v16h b[4];
#pragma unroll
      for (int kc = 0; kc < 4; ++kc)
        b[kc] = *(const v16h*)(Ufrag + nt * 2048 + (kc * 32 + lane) * 16);

      v8f acc = {};
#pragma unroll
      for (int kc = 0; kc < 4; ++kc)
        acc = __builtin_amdgcn_wmma_f32_16x16x32_f16(false, a[kc], false, b[kc],
                                                     (short)0, acc, false, false);

      const int n     = nt * 16 + (lane & 15);
      const int mBase = (lane < 16) ? 0 : 8;
#pragma unroll
      for (int r = 0; r < 8; ++r)
        zbuf[(mBase + r) * ZLD + n] = acc[r];
    }

    // ---- retire current buffer's async transfers, make tile visible -------
    if (it + 1 < SEQ) {
      asm volatile("s_wait_asynccnt 4" ::: "memory");  // keep next 4 in flight
    } else {
      asm volatile("s_wait_asynccnt 0" ::: "memory");
    }
    __syncthreads();

    // ---- gates (Keras order i,f,g,o): z = xW(staged f16) + hU -------------
    const _Float16* xs = xstage + cur * 8192 + em * 512;
#pragma unroll
    for (int e = 0; e < 8; ++e) {
      const int hc = ec0 + e;
      const float zi = zbuf[em * ZLD + hc]       + (float)xs[hc];
      const float zf = zbuf[em * ZLD + 128 + hc] + (float)xs[128 + hc];
      const float zg = zbuf[em * ZLD + 256 + hc] + (float)xs[256 + hc];
      const float zo = zbuf[em * ZLD + 384 + hc] + (float)xs[384 + hc];
      const float cc = fast_sigmoid(zf) * cst[e] + fast_sigmoid(zi) * tanhf(zg);
      const float hh = fast_sigmoid(zo) * tanhf(cc);
      cst[e] = cc;
      hst[e] = hh;
      out[((size_t)(b0 + em) * SEQ + s) * (2 * HID) + dir * HID + hc] = hh;
      hfrag[afrag_pos(em, hc)] = (_Float16)hh;
    }
    cur ^= 1;
  }

  // ---- final states: out layout = [B*T*2H] | hf | cf | hb | cb ------------
  float* fin = out + (size_t)BATCH * SEQ * 2 * HID + (size_t)dir * 2 * BATCH * HID;
  float* hF  = fin;
  float* cF  = fin + (size_t)BATCH * HID;
#pragma unroll
  for (int e = 0; e < 8; ++e) {
    const int hc = ec0 + e;
    hF[(size_t)(b0 + em) * HID + hc] = hst[e];
    cF[(size_t)(b0 + em) * HID + hc] = cst[e];
  }
}

// ===========================================================================
extern "C" void kernel_launch(void* const* d_in, const int* in_sizes, int n_in,
                              void* d_out, int out_size, void* d_ws, size_t ws_size,
                              hipStream_t stream) {
  (void)in_sizes; (void)n_in; (void)out_size; (void)ws_size;
  const int*   tokens = (const int*)d_in[0];
  const float* emb    = (const float*)d_in[1];
  const float* W_fw   = (const float*)d_in[2];
  const float* U_fw   = (const float*)d_in[3];
  const float* b_fw   = (const float*)d_in[4];
  const float* W_bw   = (const float*)d_in[5];
  const float* U_bw   = (const float*)d_in[6];
  const float* b_bw   = (const float*)d_in[7];
  float* out = (float*)d_out;

  _Float16* xWf = (_Float16*)d_ws;                       // [B*T][512] f16
  _Float16* xWb = xWf + (size_t)BATCH * SEQ * G;         // [B*T][512] f16

  // Kernel 1: fused gather + input projection for both directions.
  const size_t smem1 = (size_t)16 * 8 * 32 * 16 * 2   // W fragments
                     + (size_t)8 * 32 * 16 * 2        // A fragments
                     + 64;                            // token staging
  embed_project_kernel<<<dim3(512, 4, 1), 256, smem1, stream>>>(
      tokens, emb, W_fw, b_fw, W_bw, b_bw, xWf, xWb);

  // Kernel 2: persistent bidirectional scan (U resident in LDS,
  // async double-buffered xW staging).
  const size_t smem2 = (size_t)32 * 4 * 32 * 16 * 2   // U fragments (128 KB)
                     + (size_t)4 * 32 * 16 * 2        // h fragments
                     + (size_t)16 * ZLD * 4           // z buffer
                     + (size_t)2 * 16 * 512 * 2;      // async xW double buffer
  bilstm_scan_kernel<<<dim3(32, 1, 1), 256, smem2, stream>>>(
      xWf, xWb, U_fw, U_bw, out);
}
